// Spatial_Frequency_Attention_12214886990345
// MI455X (gfx1250) — matrix operand order repaired and checked by reading, not compiled, not run on hardware
//
#include <hip/hip_runtime.h>
#include <hip/hip_bf16.h>

typedef __attribute__((ext_vector_type(16))) _Float16 v16h;
typedef __attribute__((ext_vector_type(8)))  _Float16 v8h;
typedef __attribute__((ext_vector_type(4)))  _Float16 v4h;
typedef __attribute__((ext_vector_type(8)))  float    v8f;
typedef __attribute__((ext_vector_type(4)))  float    v4f;

#define NHEADS 6
#define HD     32
#define IMG    256
#define NB     2
#define WPL    32                 // windows per image row (256/8)
#define NWIN   (NB*WPL*WPL)       // 2048
#define DIM    192
#define NREL   225                // (2*8-1)*(2*8-1)

#define QSTR   40                 // half stride for q/k LDS rows (32 used + pad)
#define VSTR   72                 // half stride for v^T LDS rows (64 used + pad)
#define SSTR   66                 // float stride for S scratch rows
#define PSTR   72                 // half stride for P rows

__device__ __forceinline__ v16h cat16(v8h a, v8h b) {
  return __builtin_shufflevector(a, b, 0,1,2,3,4,5,6,7,8,9,10,11,12,13,14,15);
}

// ---------------- dynamic position bias MLP ----------------
__device__ __forceinline__ void mlp_layer(const float* p, const float* g, const float* be,
                                          const float* w, const float* bb, int outdim, float* out) {
  float m = 0.f;
#pragma unroll
  for (int j = 0; j < 12; ++j) m += p[j];
  m *= (1.f/12.f);
  float var = 0.f;
#pragma unroll
  for (int j = 0; j < 12; ++j) { float d = p[j]-m; var += d*d; }
  var *= (1.f/12.f);
  float inv = rsqrtf(var + 1e-5f);
  float a[12];
#pragma unroll
  for (int j = 0; j < 12; ++j) {
    float y = (p[j]-m)*inv*g[j] + be[j];
    a[j] = y > 0.f ? y : 0.f;
  }
  for (int o = 0; o < outdim; ++o) {
    float s = bb[o];
#pragma unroll
    for (int j = 0; j < 12; ++j) s += a[j]*w[j*outdim + o];
    out[o] = s;
  }
}

__global__ void dpb_mlp_kernel(const float* __restrict__ w_proj, const float* __restrict__ b_proj,
                               const float* __restrict__ g1, const float* __restrict__ be1,
                               const float* __restrict__ w1, const float* __restrict__ b1,
                               const float* __restrict__ g2, const float* __restrict__ be2,
                               const float* __restrict__ w2, const float* __restrict__ b2,
                               const float* __restrict__ g3, const float* __restrict__ be3,
                               const float* __restrict__ w3, const float* __restrict__ b3,
                               float* __restrict__ pt) {
  int t = blockIdx.x*blockDim.x + threadIdx.x;
  if (t >= NREL) return;
  float bh = (float)(t/15) - 7.f;   // 1-HSP .. HSP-1
  float bw = (float)(t%15) - 7.f;
  float p0[12];
#pragma unroll
  for (int j = 0; j < 12; ++j) p0[j] = bh*w_proj[j] + bw*w_proj[12+j] + b_proj[j];
  float p1[12]; mlp_layer(p0, g1, be1, w1, b1, 12, p1);
  float p2[12]; mlp_layer(p1, g2, be2, w2, b2, 12, p2);
  float o[6];   mlp_layer(p2, g3, be3, w3, b3, 6, o);
#pragma unroll
  for (int h = 0; h < NHEADS; ++h) pt[t*NHEADS + h] = o[h];
}

// ---------------- windowed attention, one wave32 per (window, head) ----------------
__global__ void win_attn_kernel(const float* __restrict__ qkv,
                                const float* __restrict__ pt,
                                float* __restrict__ out) {
  __shared__ alignas(16) _Float16 qs[64*QSTR];   // Q (scaled)  f16
  __shared__ alignas(16) _Float16 ks[64*QSTR];   // K           f16
  __shared__ alignas(16) _Float16 vt[HD*VSTR];   // V^T         f16
  __shared__ alignas(16) float    sb[64*SSTR];   // S scratch   f32
  __shared__ alignas(16) _Float16 ph[64*PSTR];   // P (softmax) f16

  const int lane = threadIdx.x;              // 0..31
  const int win  = blockIdx.x;               // 0..2047
  const int head = blockIdx.y;               // 0..5
  const int b  = win / (WPL*WPL);
  const int wr = win % (WPL*WPL);
  const int wh = wr / WPL, ww = wr % WPL;
  const size_t L   = (size_t)IMG*IMG;
  const size_t tsz = (size_t)NB*L*DIM;
  const float* qg = qkv + (size_t)b*L*DIM;
  const float* kg = qg + tsz;
  const float* vg = qg + 2*tsz;
  const int rowbase = (wh*8)*IMG + ww*8;
  const float scale = 0.17677669529663689f;  // 32^-0.5

  // ---- stage q,k,v for this (window, head) into LDS as f16 ----
#pragma unroll
  for (int it = 0; it < 16; ++it) {
    int flat = it*128 + lane*4;              // 64 rows x 32 ch, float4 granules
    int r = flat >> 5;
    int c = flat & 31;
    int grow = rowbase + (r>>3)*IMG + (r&7);
    size_t off = (size_t)grow*DIM + head*HD + c;
    v4f dq = *(const v4f*)(qg + off);
    v4f dk = *(const v4f*)(kg + off);
    v4f dv = *(const v4f*)(vg + off);
    v4h hq, hk;
#pragma unroll
    for (int j = 0; j < 4; ++j) { hq[j] = (_Float16)(dq[j]*scale); hk[j] = (_Float16)dk[j]; }
    *(v4h*)&qs[r*QSTR + c] = hq;
    *(v4h*)&ks[r*QSTR + c] = hk;
#pragma unroll
    for (int j = 0; j < 4; ++j) vt[(c+j)*VSTR + r] = (_Float16)dv[j];
  }
  __syncthreads();

  const int ln15 = lane & 15;
  const int hi   = lane >> 4;

  // ---- S = Q @ K^T : 4x4 tiles, one WMMA each (K=32 == head_dim) ----
#pragma unroll
  for (int i = 0; i < 4; ++i) {
    int r  = 16*i + ln15;
    int kb = hi*8;
    v8h a0 = *(const v8h*)&qs[r*QSTR + kb];
    v8h a1 = *(const v8h*)&qs[r*QSTR + kb + 16];
    v16h A = cat16(a0, a1);
#pragma unroll
    for (int j = 0; j < 4; ++j) {
      int n  = 16*j + ln15;
      int ko = hi*8;
      v8h b0 = *(const v8h*)&ks[n*QSTR + ko];
      v8h b1 = *(const v8h*)&ks[n*QSTR + ko + 16];
      v16h Bm = cat16(b0, b1);
      v8f c = {};
      c = __builtin_amdgcn_wmma_f32_16x16x32_f16(false, A, false, Bm, (short)0, c, false, false);
#pragma unroll
      for (int v = 0; v < 8; ++v)
        sb[(16*i + 8*hi + v)*SSTR + 16*j + ln15] = c[v];
    }
  }
  __syncthreads();

  // ---- bias + softmax: each lane handles 2 rows ----
#pragma unroll
  for (int rr = 0; rr < 2; ++rr) {
    int m = rr*32 + lane;
    int ihm = m >> 3, iwm = m & 7;
    float vals[64];
    float mx = -1e30f;
#pragma unroll
    for (int n = 0; n < 64; ++n) {
      int idx = (ihm - (n>>3) + 7)*15 + (iwm - (n&7) + 7);
      float v = sb[m*SSTR + n] + pt[idx*NHEADS + head];
      vals[n] = v;
      mx = fmaxf(mx, v);
    }
    float s = 0.f;
#pragma unroll
    for (int n = 0; n < 64; ++n) { float e = __expf(vals[n] - mx); vals[n] = e; s += e; }
    float inv = 1.f / s;
#pragma unroll
    for (int n = 0; n < 64; ++n) ph[m*PSTR + n] = (_Float16)(vals[n]*inv);
  }
  __syncthreads();

  // ---- O = P @ V : 4x2 tiles, 2 K-steps of 32 each ----
#pragma unroll
  for (int i = 0; i < 4; ++i) {
#pragma unroll
    for (int t = 0; t < 2; ++t) {
      v8f acc = {};
#pragma unroll
      for (int sstep = 0; sstep < 2; ++sstep) {
        int r  = 16*i + ln15;
        int kb = sstep*32 + hi*8;
        v8h a0 = *(const v8h*)&ph[r*PSTR + kb];
        v8h a1 = *(const v8h*)&ph[r*PSTR + kb + 16];
        v16h A = cat16(a0, a1);
        int n  = 16*t + ln15;
        int ko = sstep*32 + hi*8;
        v8h b0 = *(const v8h*)&vt[n*VSTR + ko];
        v8h b1 = *(const v8h*)&vt[n*VSTR + ko + 16];
        v16h Bm = cat16(b0, b1);
        acc = __builtin_amdgcn_wmma_f32_16x16x32_f16(false, A, false, Bm, (short)0, acc, false, false);
      }
#pragma unroll
      for (int v = 0; v < 8; ++v) {
        int m = 16*i + 8*hi + v;
        int grow = rowbase + (m>>3)*IMG + (m&7);
        out[((size_t)b*L + grow)*DIM + head*HD + 16*t + ln15] = acc[v];
      }
    }
  }
}

extern "C" void kernel_launch(void* const* d_in, const int* in_sizes, int n_in,
                              void* d_out, int out_size, void* d_ws, size_t ws_size,
                              hipStream_t stream) {
  const float* qkv    = (const float*)d_in[0];
  const float* w_proj = (const float*)d_in[1];
  const float* b_proj = (const float*)d_in[2];
  const float* g1  = (const float*)d_in[3];
  const float* be1 = (const float*)d_in[4];
  const float* w1  = (const float*)d_in[5];
  const float* b1  = (const float*)d_in[6];
  const float* g2  = (const float*)d_in[7];
  const float* be2 = (const float*)d_in[8];
  const float* w2  = (const float*)d_in[9];
  const float* b2  = (const float*)d_in[10];
  const float* g3  = (const float*)d_in[11];
  const float* be3 = (const float*)d_in[12];
  const float* w3  = (const float*)d_in[13];
  const float* b3  = (const float*)d_in[14];
  float* pt  = (float*)d_ws;                 // 225*6 floats
  float* out = (float*)d_out;

  dpb_mlp_kernel<<<1, 256, 0, stream>>>(w_proj, b_proj, g1, be1, w1, b1,
                                        g2, be2, w2, b2, g3, be3, w3, b3, pt);
  win_attn_kernel<<<dim3(NWIN, NHEADS), 32, 0, stream>>>(qkv, pt, out);
}
